// SpatialAwareCrossAttention_13872744366659
// MI455X (gfx1250) — matrix operand order repaired and checked by reading, not compiled, not run on hardware
//
#include <hip/hip_runtime.h>
#include <math.h>

typedef unsigned short u16;
typedef __attribute__((ext_vector_type(16))) __bf16 v16bf;
typedef __attribute__((ext_vector_type(8)))  __bf16 v8bf;
typedef __attribute__((ext_vector_type(8)))  float  v8f;

#define B_  4
#define T_  1024
#define D_  2048
#define DV_ 1024
#define P_  1024
#define H_  16
#define HD_ 128
#define SID_ 64

__device__ __forceinline__ u16 f2bf(float f) {
  unsigned u = __float_as_uint(f);
  u += 0x7FFFu + ((u >> 16) & 1u);   // round-to-nearest-even
  return (u16)(u >> 16);
}
__device__ __forceinline__ float geluf(float x) {
  return 0.5f * x * (1.0f + erff(x * 0.70710678118654752f));
}

// ---------------- elementwise f32 -> bf16 ----------------
__global__ void k_cvt_bf16(const float* __restrict__ in, u16* __restrict__ out, long n) {
  long i = (long)blockIdx.x * blockDim.x + threadIdx.x;
  long stride = (long)gridDim.x * blockDim.x;
  for (; i < n; i += stride) out[i] = f2bf(in[i]);
}

// -------- weight transpose+convert: W[K][N] f32 -> WT[N][K] bf16 --------
__global__ void k_wtrans(const float* __restrict__ in, u16* __restrict__ out, int K, int N) {
  __shared__ u16 tile[32][33];
  int k0 = blockIdx.x * 32, n0 = blockIdx.y * 32;
  int tx = threadIdx.x, ty = threadIdx.y;           // (32, 8)
  for (int r = 0; r < 32; r += 8)
    tile[ty + r][tx] = f2bf(in[(long)(k0 + ty + r) * N + n0 + tx]);
  __syncthreads();
  for (int r = 0; r < 32; r += 8)
    out[(long)(n0 + ty + r) * K + k0 + tx] = tile[tx][ty + r];
}

// -------- V transpose (bf16): per batch [P][D] -> [D][P] --------
__global__ void k_vtrans(const u16* __restrict__ in, u16* __restrict__ out) {
  __shared__ u16 tile[32][33];
  int p0 = blockIdx.x * 32, d0 = blockIdx.y * 32, b = blockIdx.z;
  const u16* ib = in + (long)b * P_ * D_;
  u16* ob = out + (long)b * D_ * P_;
  int tx = threadIdx.x, ty = threadIdx.y;
  for (int r = 0; r < 32; r += 8)
    tile[ty + r][tx] = ib[(long)(p0 + ty + r) * D_ + d0 + tx];
  __syncthreads();
  for (int r = 0; r < 32; r += 8)
    ob[(long)(d0 + ty + r) * P_ + p0 + tx] = tile[tx][ty + r];
}

// ---------------- bf16 WMMA GEMM: C = A(MxK) * Bt(NxK)^T + bias ----------------
// Block tile 128x128, K-step 32, DOUBLE-BUFFERED LDS (one barrier per step).
// 8 waves: 4 along M (32 rows each), 2 along N (64 cols). Per wave: 2x4 accum
// tiles of 16x16 -> 64 v_wmma_f32_16x16x32_bf16 per block K-step.
__global__ __launch_bounds__(256) void k_gemm_bf16(
    const u16* __restrict__ A, int lda, long sAo, long sAi,
    const u16* __restrict__ Bt, int ldb, long sBo, long sBi,
    void* __restrict__ Cv, int ldc, long sCo, long sCi,
    int K, int inner_count, const float* __restrict__ bias, int store_bf16) {
  __shared__ __align__(16) u16 lA[2][128 * 40];
  __shared__ __align__(16) u16 lB[2][128 * 40];
  int z = blockIdx.z;
  int inner = z % inner_count;
  int outer = z / inner_count;
  A  += outer * sAo + inner * sAi;
  Bt += outer * sBo + inner * sBi;
  long coff = outer * sCo + inner * sCi;

  const int bm = blockIdx.x * 128;
  const int bn = blockIdx.y * 128;
  const int t = threadIdx.x;
  const int lane = t & 31;
  const int w = t >> 5;
  const int wm = (w & 3) * 32;   // wave row offset in tile
  const int wn = (w >> 2) * 64;  // wave col offset in tile
  const int fl = lane & 15;
  const int fh = lane >> 4;

  v8f zero8 = {0.f, 0.f, 0.f, 0.f, 0.f, 0.f, 0.f, 0.f};
  v8f acc[2][4];
#pragma unroll
  for (int i = 0; i < 2; i++)
#pragma unroll
    for (int j = 0; j < 4; j++) acc[i][j] = zero8;

  const int lrow = t >> 1;         // 0..127
  const int lseg = (t & 1) * 16;   // element offset {0,16}
  const u16* gArow = A + (long)(bm + lrow) * lda + lseg;
  const u16* gBrow = Bt + (long)(bn + lrow) * ldb + lseg;
  const int ldsoff = lrow * 40 + lseg;

  // prologue: stage step 0 into buffer 0
  {
    uint4 pa0 = *(const uint4*)(gArow);
    uint4 pa1 = *(const uint4*)(gArow + 8);
    uint4 pb0 = *(const uint4*)(gBrow);
    uint4 pb1 = *(const uint4*)(gBrow + 8);
    *(uint4*)&lA[0][ldsoff]     = pa0;
    *(uint4*)&lA[0][ldsoff + 8] = pa1;
    *(uint4*)&lB[0][ldsoff]     = pb0;
    *(uint4*)&lB[0][ldsoff + 8] = pb1;
  }
  __syncthreads();

  const int nsteps = K >> 5;
  int cur = 0;
  for (int s = 0; s < nsteps; ++s) {
    // issue global loads for step s+1 early (hidden behind WMMA stage)
    uint4 na0 = {0, 0, 0, 0}, na1 = {0, 0, 0, 0};
    uint4 nb0 = {0, 0, 0, 0}, nb1 = {0, 0, 0, 0};
    const bool more = (s + 1) < nsteps;
    if (more) {
      const u16* pA = gArow + (long)(s + 1) * 32;
      const u16* pB = gBrow + (long)(s + 1) * 32;
      na0 = *(const uint4*)(pA);
      na1 = *(const uint4*)(pA + 8);
      nb0 = *(const uint4*)(pB);
      nb1 = *(const uint4*)(pB + 8);
      if (s + 2 < nsteps) {  // pull step s+2 lines toward L2 (global_prefetch_b8)
        __builtin_prefetch(gArow + (long)(s + 2) * 32, 0, 0);
        __builtin_prefetch(gBrow + (long)(s + 2) * 32, 0, 0);
      }
    }

    // compute on buffer `cur`
    const u16* la = lA[cur];
    const u16* lb = lB[cur];
    v16bf av[2], bv4[4];
#pragma unroll
    for (int i = 0; i < 2; i++) {
      // A 16x32 bf16 layout: lane<16 -> K 0..7 & 16..23; lane>=16 -> K 8..15 & 24..31
      const u16* base = &la[(wm + i * 16 + fl) * 40 + fh * 8];
      v8bf lo = *(const v8bf*)(base);
      v8bf hi = *(const v8bf*)(base + 16);
      av[i] = __builtin_shufflevector(lo, hi, 0,1,2,3,4,5,6,7,8,9,10,11,12,13,14,15);
    }
#pragma unroll
    for (int j = 0; j < 4; j++) {
      // B 32x16 bf16 layout: lane<16 -> K 0..7 & 16..23 at N=lane; lane>=16 -> K 8..15 & 24..31
      const u16* base = &lb[(wn + j * 16 + fl) * 40 + fh * 8];
      v8bf lo = *(const v8bf*)(base);
      v8bf hi = *(const v8bf*)(base + 16);
      bv4[j] = __builtin_shufflevector(lo, hi, 0,1,2,3,4,5,6,7,8,9,10,11,12,13,14,15);
    }
#pragma unroll
    for (int i = 0; i < 2; i++)
#pragma unroll
      for (int j = 0; j < 4; j++)
        acc[i][j] = __builtin_amdgcn_wmma_f32_16x16x32_bf16(
            false, av[i], false, bv4[j], (short)0, acc[i][j], false, false);

    // stage step s+1 into the other buffer; single barrier per step
    if (more) {
      u16* da = lA[cur ^ 1];
      u16* db = lB[cur ^ 1];
      *(uint4*)&da[ldsoff]     = na0;
      *(uint4*)&da[ldsoff + 8] = na1;
      *(uint4*)&db[ldsoff]     = nb0;
      *(uint4*)&db[ldsoff + 8] = nb1;
      __syncthreads();
    }
    cur ^= 1;
  }

  // epilogue: C layout — lane&15 = col, lane>>4 selects M 0..7 / 8..15 halves
#pragma unroll
  for (int i = 0; i < 2; i++) {
#pragma unroll
    for (int j = 0; j < 4; j++) {
      int col = bn + wn + j * 16 + fl;
      int row0 = bm + wm + i * 16 + fh * 8;
      float bb = bias ? bias[col] : 0.0f;
      if (store_bf16) {
        u16* C = (u16*)Cv;
#pragma unroll
        for (int v = 0; v < 8; v++)
          C[coff + (long)(row0 + v) * ldc + col] = f2bf(acc[i][j][v] + bb);
      } else {
        float* C = (float*)Cv;
#pragma unroll
        for (int v = 0; v < 8; v++)
          C[coff + (long)(row0 + v) * ldc + col] = acc[i][j][v] + bb;
      }
    }
  }
}

// ---------------- generic in-place row softmax (n multiple of 256) ----------------
__global__ __launch_bounds__(256) void k_row_softmax(float* __restrict__ x, int n) {
  __shared__ float red[256];
  float* row = x + (long)blockIdx.x * n;
  int t = threadIdx.x;
  float m = -3.0e38f;
  for (int i = t; i < n; i += 256) m = fmaxf(m, row[i]);
  red[t] = m; __syncthreads();
  for (int s = 128; s > 0; s >>= 1) { if (t < s) red[t] = fmaxf(red[t], red[t + s]); __syncthreads(); }
  m = red[0]; __syncthreads();
  float sum = 0.f;
  for (int i = t; i < n; i += 256) { float e = expf(row[i] - m); row[i] = e; sum += e; }
  red[t] = sum; __syncthreads();
  for (int s = 128; s > 0; s >>= 1) { if (t < s) red[t] += red[t + s]; __syncthreads(); }
  float rs = 1.0f / red[0];
  for (int i = t; i < n; i += 256) row[i] *= rs;
}

// ------ attention softmax: scale + spatial bias, in-place f32, plus bf16 copy ------
__global__ __launch_bounds__(256) void k_attn_softmax(float* __restrict__ attn, u16* __restrict__ attn_bf,
                                                      const float* __restrict__ sbias, int ht) {
  __shared__ float red[256];
  long row = blockIdx.x;
  int b = (int)(row / ht);
  float* x = attn + row * 1024;
  const float* sb = sbias + b * 1024;
  int t = threadIdx.x;
  const float inv = 0.08838834764831845f;  // 1/sqrt(128)
  float vals[4]; float m = -3.0e38f;
#pragma unroll
  for (int i = 0; i < 4; i++) { int p = t + i * 256; float v = x[p] * inv + sb[p]; vals[i] = v; m = fmaxf(m, v); }
  red[t] = m; __syncthreads();
  for (int s = 128; s > 0; s >>= 1) { if (t < s) red[t] = fmaxf(red[t], red[t + s]); __syncthreads(); }
  m = red[0]; __syncthreads();
  float sum = 0.f;
#pragma unroll
  for (int i = 0; i < 4; i++) { vals[i] = expf(vals[i] - m); sum += vals[i]; }
  red[t] = sum; __syncthreads();
  for (int s = 128; s > 0; s >>= 1) { if (t < s) red[t] += red[t + s]; __syncthreads(); }
  float rs = 1.0f / red[0];
#pragma unroll
  for (int i = 0; i < 4; i++) {
    int p = t + i * 256; float a = vals[i] * rs;
    x[p] = a; attn_bf[row * 1024 + p] = f2bf(a);
  }
}

// ---------------- residual + LayerNorm ----------------
__global__ __launch_bounds__(256) void k_ln(const float* __restrict__ pre, const float* __restrict__ resid,
                                            const float* __restrict__ g, const float* __restrict__ be,
                                            float* __restrict__ out) {
  __shared__ float red[256];
  long row = blockIdx.x;
  const float* pr = pre + row * D_;
  const float* rr = resid + row * D_;
  float* o = out + row * D_;
  int t = threadIdx.x;
  float xv[8]; float s = 0.f, s2 = 0.f;
#pragma unroll
  for (int i = 0; i < 8; i++) { int c = t + i * 256; float v = pr[c] + rr[c]; xv[i] = v; s += v; s2 += v * v; }
  red[t] = s; __syncthreads();
  for (int k = 128; k > 0; k >>= 1) { if (t < k) red[t] += red[t + k]; __syncthreads(); }
  s = red[0]; __syncthreads();
  red[t] = s2; __syncthreads();
  for (int k = 128; k > 0; k >>= 1) { if (t < k) red[t] += red[t + k]; __syncthreads(); }
  s2 = red[0];
  float mu = s * (1.0f / D_);
  float var = s2 * (1.0f / D_) - mu * mu;
  float r = rsqrtf(var + 1e-5f);
#pragma unroll
  for (int i = 0; i < 8; i++) { int c = t + i * 256; o[c] = (xv[i] - mu) * r * g[c] + be[c]; }
}

// ---------------- spatial intent pipeline (algebraically folded) ----------------
__global__ void k_mq(const float* __restrict__ sq, const float* __restrict__ mWq,
                     const float* __restrict__ mbq, float* __restrict__ mq) {
  int n = blockIdx.x * 256 + threadIdx.x;
  float s = mbq[n];
  for (int i = 0; i < D_; i++) s += sq[i] * mWq[(long)i * D_ + n];
  mq[n] = s;
}
__global__ void k_c(const float* __restrict__ mbk, const float* __restrict__ mq, float* __restrict__ cvec) {
  __shared__ float red[256];
  int h = blockIdx.x; int t = threadIdx.x;
  float s = 0.f;
  for (int j = t; j < 512; j += 256) s += mbk[h * 512 + j] * mq[h * 512 + j];
  red[t] = s; __syncthreads();
  for (int k = 128; k > 0; k >>= 1) { if (t < k) red[t] += red[t + k]; __syncthreads(); }
  if (t == 0) cvec[h] = red[0];
}
__global__ void k_u(const float* __restrict__ mWk, const float* __restrict__ mq, float* __restrict__ u) {
  int i = blockIdx.x * 256 + threadIdx.x;
  int h = blockIdx.y;
  const float* wr = mWk + (long)i * D_ + h * 512;
  const float* qr = mq + h * 512;
  float s = 0.f;
  for (int j = 0; j < 512; j++) s += wr[j] * qr[j];
  u[h * D_ + i] = s;
}
__global__ __launch_bounds__(256) void k_ms(const float* __restrict__ text, const float* __restrict__ u,
                                            const float* __restrict__ cvec, float* __restrict__ ms) {
  __shared__ float red[256];
  int t = threadIdx.x;
  const float* trow = text + (long)blockIdx.x * D_;
  float a[4] = {0.f, 0.f, 0.f, 0.f};
  for (int i = t; i < D_; i += 256) {
    float xv = trow[i];
    a[0] += xv * u[i]; a[1] += xv * u[D_ + i]; a[2] += xv * u[2 * D_ + i]; a[3] += xv * u[3 * D_ + i];
  }
  int b = blockIdx.x >> 10, tt = blockIdx.x & 1023;
  for (int h = 0; h < 4; h++) {
    red[t] = a[h]; __syncthreads();
    for (int k = 128; k > 0; k >>= 1) { if (t < k) red[t] += red[t + k]; __syncthreads(); }
    if (t == 0) ms[((b * 4 + h) << 10) + tt] = (red[0] + cvec[h]) * 0.04419417382415922f; // 1/sqrt(512)
    __syncthreads();
  }
}
__global__ void k_wtext(const float* __restrict__ mw, const float* __restrict__ text, float* __restrict__ wtext) {
  int i = blockIdx.x * 256 + threadIdx.x;
  int h = blockIdx.y, b = blockIdx.z;
  const float* wrow = mw + ((long)(b * 4 + h) << 10);
  const float* tb = text + (long)b * T_ * D_ + i;
  float s = 0.f;
  for (int t2 = 0; t2 < T_; t2++) s += wrow[t2] * tb[(long)t2 * D_];
  wtext[((long)(b * 4 + h)) * D_ + i] = s;
}
__global__ void k_agg(const float* __restrict__ wtext, const float* __restrict__ mWv,
                      const float* __restrict__ mbv, float* __restrict__ agg) {
  int n = blockIdx.x * 256 + threadIdx.x;
  int b = blockIdx.y;
  int h = n >> 9;
  const float* wt = wtext + ((long)(b * 4 + h)) * D_;
  float s = mbv[n];
  for (int i = 0; i < D_; i++) s += wt[i] * mWv[(long)i * D_ + n];
  agg[(long)b * D_ + n] = s;
}
__global__ void k_vecmat(const float* __restrict__ vin, const float* __restrict__ W,
                         const float* __restrict__ bias, float* __restrict__ out,
                         int Kd, int Nd, int do_gelu) {
  int n = blockIdx.x * 256 + threadIdx.x;
  int b = blockIdx.y;
  if (n >= Nd) return;
  float s = bias[n];
  const float* v = vin + (long)b * Kd;
  for (int i = 0; i < Kd; i++) s += v[i] * W[(long)i * Nd + n];
  if (do_gelu) s = geluf(s);
  out[(long)b * Nd + n] = s;
}
__global__ void k_pe(const float* __restrict__ pe1w, const float* __restrict__ pe1b,
                     const float* __restrict__ pe2w, const float* __restrict__ pe2b,
                     float* __restrict__ pe) {
  int p = blockIdx.x * 256 + threadIdx.x;
  float r = (float)(p >> 5) * (1.0f / 31.0f);
  float c = (float)(p & 31) * (1.0f / 31.0f);
  float h[32];
  for (int j = 0; j < 32; j++) h[j] = geluf(r * pe1w[j] + c * pe1w[32 + j] + pe1b[j]);
  for (int s = 0; s < SID_; s++) {
    float v = pe2b[s];
    for (int j = 0; j < 32; j++) v += h[j] * pe2w[j * SID_ + s];
    pe[(long)p * SID_ + s] = v;
  }
}
__global__ void k_sbias(const float* __restrict__ intent, const float* __restrict__ pe,
                        const float* __restrict__ ml1w, const float* __restrict__ ml1b,
                        const float* __restrict__ ml2w, const float* __restrict__ ml2b,
                        const float* __restrict__ bscale, float* __restrict__ sbias) {
  int p = blockIdx.x * 256 + threadIdx.x;
  int b = blockIdx.y;
  float it[SID_], pv[SID_];
  for (int j = 0; j < SID_; j++) { it[j] = intent[b * SID_ + j]; pv[j] = pe[(long)p * SID_ + j]; }
  float acc = ml2b[0];
  for (int s = 0; s < SID_; s++) {
    float hv = ml1b[s];
    for (int j = 0; j < SID_; j++)
      hv += it[j] * ml1w[j * SID_ + s] + pv[j] * ml1w[(SID_ + j) * SID_ + s];
    acc += geluf(hv) * ml2w[s];
  }
  sbias[(long)b * P_ + p] = (*bscale) * acc;  // bias_scale folded here
}

extern "C" void kernel_launch(void* const* d_in, const int* in_sizes, int n_in,
                              void* d_out, int out_size, void* d_ws, size_t ws_size,
                              hipStream_t stream) {
  (void)in_sizes; (void)n_in; (void)out_size; (void)ws_size;
  const float* text = (const float*)d_in[0];
  const float* vis  = (const float*)d_in[1];
  const float* Wq = (const float*)d_in[4];  const float* bq  = (const float*)d_in[5];
  const float* Wk = (const float*)d_in[6];  const float* bk  = (const float*)d_in[7];
  const float* Wv = (const float*)d_in[8];  const float* bvv = (const float*)d_in[9];
  const float* Wo = (const float*)d_in[10]; const float* bo  = (const float*)d_in[11];
  const float* sq = (const float*)d_in[12];
  const float* mWq = (const float*)d_in[13]; const float* mbq = (const float*)d_in[14];
  const float* mWk = (const float*)d_in[15]; const float* mbk = (const float*)d_in[16];
  const float* mWv = (const float*)d_in[17]; const float* mbv = (const float*)d_in[18];
  const float* mWo = (const float*)d_in[19]; const float* mbo = (const float*)d_in[20];
  const float* ip1w = (const float*)d_in[21]; const float* ip1b = (const float*)d_in[22];
  const float* ip2w = (const float*)d_in[23]; const float* ip2b = (const float*)d_in[24];
  const float* pe1w = (const float*)d_in[25]; const float* pe1b = (const float*)d_in[26];
  const float* pe2w = (const float*)d_in[27]; const float* pe2b = (const float*)d_in[28];
  const float* ml1w = (const float*)d_in[29]; const float* ml1b = (const float*)d_in[30];
  const float* ml2w = (const float*)d_in[31]; const float* ml2b = (const float*)d_in[32];
  const float* bscale = (const float*)d_in[33];
  const float* lng = (const float*)d_in[34]; const float* lnb = (const float*)d_in[35];

  // ---- workspace carve-out (~303 MB) ----
  char* ws = (char*)d_ws;
  size_t off = 0;
  auto alloc = [&](size_t bytes) -> char* {
    char* p = ws + off; off += (bytes + 255) & ~(size_t)255; return p;
  };
  u16* textbf = (u16*)alloc((size_t)B_ * T_ * D_ * 2);
  u16* visbf  = (u16*)alloc((size_t)B_ * P_ * DV_ * 2);
  u16* WqT = (u16*)alloc((size_t)D_ * D_ * 2);
  u16* WkT = (u16*)alloc((size_t)D_ * DV_ * 2);
  u16* WvT = (u16*)alloc((size_t)D_ * DV_ * 2);
  u16* WoT = (u16*)alloc((size_t)D_ * D_ * 2);
  u16* Qbf = (u16*)alloc((size_t)B_ * T_ * D_ * 2);
  u16* Kbf = (u16*)alloc((size_t)B_ * P_ * D_ * 2);
  u16* Vbf = (u16*)alloc((size_t)B_ * P_ * D_ * 2);
  u16* Vt  = (u16*)alloc((size_t)B_ * D_ * P_ * 2);
  u16* attnbf = (u16*)alloc((size_t)B_ * H_ * T_ * P_ * 2);
  u16* ctxbf  = (u16*)alloc((size_t)B_ * T_ * D_ * 2);
  float* outpre = (float*)alloc((size_t)B_ * T_ * D_ * 4);
  float* mq    = (float*)alloc((size_t)D_ * 4);
  float* cvec  = (float*)alloc(4 * 4);
  float* u     = (float*)alloc((size_t)4 * D_ * 4);
  float* ms    = (float*)alloc((size_t)B_ * 4 * T_ * 4);
  float* wtext = (float*)alloc((size_t)B_ * 4 * D_ * 4);
  float* agg   = (float*)alloc((size_t)B_ * D_ * 4);
  float* agg2  = (float*)alloc((size_t)B_ * D_ * 4);
  float* h1    = (float*)alloc((size_t)B_ * 1024 * 4);
  float* intent = (float*)alloc((size_t)B_ * SID_ * 4);
  float* pe    = (float*)alloc((size_t)P_ * SID_ * 4);
  float* sbias = (float*)alloc((size_t)B_ * P_ * 4);

  float* out_f  = (float*)d_out;                      // (B,T,D)
  float* attn_f = out_f + (size_t)B_ * T_ * D_;       // (B,H,T,P)

  dim3 b256(256);
  dim3 tb(32, 8);

  // conversions / transposes
  k_cvt_bf16<<<4096, b256, 0, stream>>>(text, textbf, (long)B_ * T_ * D_);
  k_cvt_bf16<<<4096, b256, 0, stream>>>(vis, visbf, (long)B_ * P_ * DV_);
  k_wtrans<<<dim3(64, 64), tb, 0, stream>>>(Wq, WqT, D_, D_);
  k_wtrans<<<dim3(32, 64), tb, 0, stream>>>(Wk, WkT, DV_, D_);
  k_wtrans<<<dim3(32, 64), tb, 0, stream>>>(Wv, WvT, DV_, D_);
  k_wtrans<<<dim3(64, 64), tb, 0, stream>>>(Wo, WoT, D_, D_);

  // Q/K/V projections (bf16 WMMA)
  k_gemm_bf16<<<dim3(32, 16, 1), b256, 0, stream>>>(textbf, D_, 0, 0, WqT, D_, 0, 0,
      Qbf, D_, 0, 0, D_, 1, bq, 1);
  k_gemm_bf16<<<dim3(32, 16, 1), b256, 0, stream>>>(visbf, DV_, 0, 0, WkT, DV_, 0, 0,
      Kbf, D_, 0, 0, DV_, 1, bk, 1);
  k_gemm_bf16<<<dim3(32, 16, 1), b256, 0, stream>>>(visbf, DV_, 0, 0, WvT, DV_, 0, 0,
      Vbf, D_, 0, 0, DV_, 1, bvv, 1);
  k_vtrans<<<dim3(32, 64, B_), tb, 0, stream>>>(Vbf, Vt);

  // spatial intent pipeline (folded; tiny)
  k_mq<<<8, b256, 0, stream>>>(sq, mWq, mbq, mq);
  k_c<<<4, b256, 0, stream>>>(mbk, mq, cvec);
  k_u<<<dim3(8, 4), b256, 0, stream>>>(mWk, mq, u);
  k_ms<<<B_ * T_, b256, 0, stream>>>(text, u, cvec, ms);
  k_row_softmax<<<16, b256, 0, stream>>>(ms, 1024);
  k_wtext<<<dim3(8, 4, 4), b256, 0, stream>>>(ms, text, wtext);
  k_agg<<<dim3(8, 4), b256, 0, stream>>>(wtext, mWv, mbv, agg);
  k_vecmat<<<dim3(8, 4), b256, 0, stream>>>(agg, mWo, mbo, agg2, D_, D_, 0);
  k_vecmat<<<dim3(4, 4), b256, 0, stream>>>(agg2, ip1w, ip1b, h1, D_, 1024, 1);
  k_vecmat<<<dim3(1, 4), b256, 0, stream>>>(h1, ip2w, ip2b, intent, 1024, SID_, 0);
  k_pe<<<4, b256, 0, stream>>>(pe1w, pe1b, pe2w, pe2b, pe);
  k_sbias<<<dim3(4, 4), b256, 0, stream>>>(intent, pe, ml1w, ml1b, ml2w, ml2b, bscale, sbias);

  // scores = Q_bh @ K_bh^T -> f32 directly into attn output region (batched over z=b*16+h)
  k_gemm_bf16<<<dim3(8, 8, 64), b256, 0, stream>>>(
      Qbf, D_, (long)T_ * D_, 128,
      Kbf, D_, (long)P_ * D_, 128,
      attn_f, P_, (long)16 * T_ * P_, (long)T_ * P_,
      HD_, 16, nullptr, 0);

  // biased softmax in place (f32) + bf16 copy for ctx GEMM
  k_attn_softmax<<<B_ * H_ * T_, b256, 0, stream>>>(attn_f, attnbf, sbias, H_ * T_);

  // ctx = attn @ V  (batched; Vt used as N-major operand)
  k_gemm_bf16<<<dim3(8, 1, 64), b256, 0, stream>>>(
      attnbf, P_, (long)16 * T_ * P_, (long)T_ * P_,
      Vt, P_, (long)D_ * P_, (long)128 * P_,
      ctxbf, D_, (long)T_ * D_, 128,
      P_, 16, nullptr, 1);

  // out projection + bias (f32 out)
  k_gemm_bf16<<<dim3(32, 16, 1), b256, 0, stream>>>(ctxbf, D_, 0, 0, WoT, D_, 0, 0,
      outpre, D_, 0, 0, D_, 1, bo, 0);

  // residual + LayerNorm
  k_ln<<<B_ * T_, b256, 0, stream>>>(outpre, text, lng, lnb, out_f);
}
